// GRACE_90589450207541
// MI455X (gfx1250) — matrix operand order repaired and checked by reading, not compiled, not run on hardware
//
#include <hip/hip_runtime.h>
#include <hip/hip_bf16.h>

// GCN forward for MI455X (gfx1250, wave32, WMMA).
//
// Roofline: adj + aug_adj1 = 512 MB f32 streamed once from HBM (~23us @23.3TB/s).
// Math = ~71 GFLOP -> AI ~130 FLOP/B ~= bf16 machine balance, so both the
// streaming path and the WMMA path matter. bf16 WMMA (16x16x32, f32 accum),
// packed v_cvt_pk_bf16_f32 conversions, NT hints on the single-pass adj
// stream. GEMM2: 128 WGs x 8 waves, low register pressure for occupancy
// (latency hiding), WG tile 64x256 covers all output columns -> adj read once.

typedef __attribute__((ext_vector_type(16))) __bf16 v16bf;
typedef __attribute__((ext_vector_type(8)))  float  v8f;
typedef __attribute__((ext_vector_type(4)))  float  f32x4;
typedef __attribute__((ext_vector_type(4)))  __bf16 bf16x4;
typedef __attribute__((ext_vector_type(8)))  __bf16 bf16x8;
typedef __attribute__((ext_vector_type(4)))  unsigned int u32x4;

#define N_ROWS 8192
#define IN_FT  512
#define OUT_FT 256

union BF16Frag {
  v16bf v;
  bf16x4 h[4];
  u32x4 q[2];
};

// Load a 16x32 bf16 A-fragment from an f32 row-major matrix (non-temporal).
// 16-bit A layout: lane half 0 holds K 0-7 & 16-23, half 1 holds K 8-15 & 24-31.
__device__ __forceinline__ BF16Frag load_a_frag_nt(const float* rowBase, int kA) {
  const f32x4* p0 = reinterpret_cast<const f32x4*>(rowBase + kA);
  const f32x4* p1 = reinterpret_cast<const f32x4*>(rowBase + kA + 16);
  f32x4 x0 = __builtin_nontemporal_load(p0);
  f32x4 x1 = __builtin_nontemporal_load(p0 + 1);
  f32x4 x2 = __builtin_nontemporal_load(p1);
  f32x4 x3 = __builtin_nontemporal_load(p1 + 1);
  BF16Frag f;
  f.h[0] = __builtin_convertvector(x0, bf16x4);
  f.h[1] = __builtin_convertvector(x1, bf16x4);
  f.h[2] = __builtin_convertvector(x2, bf16x4);
  f.h[3] = __builtin_convertvector(x3, bf16x4);
  return f;
}

// ---------------------------------------------------------------------------
// Kernel 1: seqT[o][k] = sum_i features[k][i] * W[o][i]   (bf16 output, 4 MB)
// Block = 256 threads = 8 waves (4 across M, 2 across N). WG tile 64 x 256.
// ---------------------------------------------------------------------------
__global__ __launch_bounds__(256) void gcn_gemm1_featW(
    const float* __restrict__ feat,          // [8192, 512]
    const float* __restrict__ W,             // [256, 512]
    unsigned short* __restrict__ seqT)       // [256, 8192] bf16 bits
{
  const int tid  = threadIdx.x;
  const int wave = tid >> 5;
  const int lane = tid & 31;
  const int wm = wave & 3;         // 0..3  -> M sub-tile
  const int wn = wave >> 2;        // 0..1  -> N half
  const int lr = lane & 15;
  const int lh = lane >> 4;        // lane half

  const int m_base = blockIdx.x * 64 + wm * 16;
  const int o_base = wn * 128;

  v8f acc[8] = {};

  const float* aRow = feat + (size_t)(m_base + lr) * IN_FT;

  for (int k0 = 0; k0 < IN_FT; k0 += 32) {
    BF16Frag aF;
    {
      const int kA = k0 + lh * 8;
      const f32x4* p0 = reinterpret_cast<const f32x4*>(aRow + kA);
      const f32x4* p1 = reinterpret_cast<const f32x4*>(aRow + kA + 16);
      aF.h[0] = __builtin_convertvector(p0[0], bf16x4);
      aF.h[1] = __builtin_convertvector(p0[1], bf16x4);
      aF.h[2] = __builtin_convertvector(p1[0], bf16x4);
      aF.h[3] = __builtin_convertvector(p1[1], bf16x4);
    }
#pragma unroll
    for (int t = 0; t < 8; ++t) {
      // B fragment: column o = W row o (contiguous in K).
      const int o  = o_base + 16 * t + lr;
      const int kB = k0 + lh * 8;
      const f32x4* pb = reinterpret_cast<const f32x4*>(W + (size_t)o * IN_FT + kB);
      BF16Frag bF;
      bF.h[0] = __builtin_convertvector(pb[0], bf16x4);
      bF.h[1] = __builtin_convertvector(pb[1], bf16x4);
      bF.h[2] = __builtin_convertvector(pb[4], bf16x4);
      bF.h[3] = __builtin_convertvector(pb[5], bf16x4);
      acc[t] = __builtin_amdgcn_wmma_f32_16x16x32_bf16(
          false, aF.v, false, bF.v, (short)0, acc[t], false, false);
    }
  }

  // Transposed bf16 store: seqT[o][m]. Scattered b16 stores, but total is 4 MB.
#pragma unroll
  for (int t = 0; t < 8; ++t) {
    const int o = o_base + 16 * t + lr;
    union { bf16x8 h; unsigned short s[8]; } cv;
    cv.h = __builtin_convertvector(acc[t], bf16x8);
#pragma unroll
    for (int r = 0; r < 8; ++r) {
      const int m = m_base + r + 8 * lh;   // 32-bit C/D layout
      seqT[(size_t)o * N_ROWS + m] = cv.s[r];
    }
  }
}

// ---------------------------------------------------------------------------
// Kernel 2: out[n][o] = prelu( sum_m adj[n][m] * seqT[o][m] + bias[o] )
// WG tile 64 x 256, 8 waves (2 M x 4 N), wave tile 32 x 64 (2 M x 4 N tiles).
// Grid = 128 WGs. Each adj element fetched once from HBM (NT); each B load
// feeds 2 WMMAs. Low register pressure -> high occupancy -> HBM latency hiding.
// ---------------------------------------------------------------------------
__global__ __launch_bounds__(256) void gcn_gemm2_aggregate(
    const float* __restrict__ adj,                  // [8192, 8192]
    const unsigned short* __restrict__ seqT,        // [256, 8192] bf16
    const float* __restrict__ bias,                 // [256]
    const float* __restrict__ prelu_a,              // [1]
    float* __restrict__ out)                        // [8192, 256]
{
  const int tid  = threadIdx.x;
  const int wave = tid >> 5;
  const int lane = tid & 31;
  const int wm = wave & 1;         // 0..1 -> M group (two 16-row tiles each)
  const int wn = wave >> 1;        // 0..3 -> N group (four 16-col tiles)
  const int lr = lane & 15;
  const int lh = lane >> 4;

  const int m_base = blockIdx.x * 64 + wm * 32;
  const int o_base = wn * 64;

  v8f acc0[4] = {};
  v8f acc1[4] = {};

  const float* aRow0 = adj + (size_t)(m_base + lr) * N_ROWS;
  const float* aRow1 = aRow0 + (size_t)16 * N_ROWS;

#pragma unroll 2
  for (int k0 = 0; k0 < N_ROWS; k0 += 32) {
    const int kA = k0 + lh * 8;
    BF16Frag aF0 = load_a_frag_nt(aRow0, kA);
    BF16Frag aF1 = load_a_frag_nt(aRow1, kA);
#pragma unroll
    for (int t = 0; t < 4; ++t) {
      // B fragment straight from bf16 seqT: 2 x 16B per lane (K 8h..8h+7 and
      // 16+8h..16+8h+7), matching the A fragment's K order.
      const int o  = o_base + 16 * t + lr;
      const int kB = k0 + lh * 8;
      const u32x4* pq =
          reinterpret_cast<const u32x4*>(seqT + (size_t)o * N_ROWS + kB);
      BF16Frag bF;
      bF.q[0] = pq[0];
      bF.q[1] = pq[2];
      acc0[t] = __builtin_amdgcn_wmma_f32_16x16x32_bf16(
          false, aF0.v, false, bF.v, (short)0, acc0[t], false, false);
      acc1[t] = __builtin_amdgcn_wmma_f32_16x16x32_bf16(
          false, aF1.v, false, bF.v, (short)0, acc1[t], false, false);
    }
  }

  const float slope = prelu_a[0];
#pragma unroll
  for (int t = 0; t < 4; ++t) {
    const int o = o_base + 16 * t + lr;
    const float b = bias[o];
#pragma unroll
    for (int r = 0; r < 8; ++r) {
      const int m0 = m_base + r + 8 * lh;
      float v0 = acc0[t][r] + b;
      v0 = (v0 >= 0.0f) ? v0 : slope * v0;
      out[(size_t)m0 * OUT_FT + o] = v0;
      float v1 = acc1[t][r] + b;
      v1 = (v1 >= 0.0f) ? v1 : slope * v1;
      out[(size_t)(m0 + 16) * OUT_FT + o] = v1;
    }
  }
}

// ---------------------------------------------------------------------------
extern "C" void kernel_launch(void* const* d_in, const int* in_sizes, int n_in,
                              void* d_out, int out_size, void* d_ws, size_t ws_size,
                              hipStream_t stream) {
  const float* feat = (const float*)d_in[0];   // [1,8192,512]
  const float* adj  = (const float*)d_in[1];   // [1,8192,8192]
  const float* aug  = (const float*)d_in[2];   // [1,8192,8192]
  const float* W    = (const float*)d_in[3];   // [256,512]
  const float* bias = (const float*)d_in[4];   // [256]
  const float* pa   = (const float*)d_in[5];   // [1]
  (void)in_sizes; (void)n_in; (void)out_size; (void)ws_size;

  float* out = (float*)d_out;                       // h_0 || h_1
  unsigned short* seqT = (unsigned short*)d_ws;     // [256,8192] bf16, 4 MB

  dim3 blk(256);

  gcn_gemm1_featW<<<dim3(N_ROWS / 64), blk, 0, stream>>>(feat, W, seqT);
  gcn_gemm2_aggregate<<<dim3(N_ROWS / 64), blk, 0, stream>>>(
      adj, seqT, bias, pa, out);
  gcn_gemm2_aggregate<<<dim3(N_ROWS / 64), blk, 0, stream>>>(
      aug, seqT, bias, pa, out + (size_t)N_ROWS * OUT_FT);
}
